// MultiHeadAttention_14894946582725
// MI455X (gfx1250) — matrix operand order repaired and checked by reading, not compiled, not run on hardware
//
#include <hip/hip_runtime.h>
#include <hip/hip_bf16.h>

// ---------------------------------------------------------------------------
// MHA for MI455X (gfx1250): bf16 WMMA 16x16x32 with f32 accumulate.
// B=4, S=8192, D=128, H=8, Sp=1024.
// d_out = [ output f32 (B*S*D) | attention f32 (B*H*Sp*Sp) ]
// Shared B tiles are staged in LDS (padded ld=136 -> conflict-free b128
// reads) via GLOBAL_LOAD_ASYNC_TO_LDS when the toolchain exposes it.
// ---------------------------------------------------------------------------

#define BB   4
#define SS   8192
#define DD   128
#define HH   8
#define SP   1024
#define LDP  136   // padded LDS leading dim (bf16 elements): 68 dwords/row

typedef __attribute__((ext_vector_type(16))) __bf16 v16bf;
typedef __attribute__((ext_vector_type(8)))  float  v8f;
typedef __attribute__((ext_vector_type(4)))  float  f32x4;
typedef __attribute__((ext_vector_type(4)))  int    i32x4;

#define AS1 __attribute__((address_space(1)))
#define AS3 __attribute__((address_space(3)))

#if defined(__AMDGCN__) && __has_builtin(__builtin_amdgcn_global_load_async_to_lds_b128) && __has_builtin(__builtin_amdgcn_s_wait_asynccnt)
#define HAVE_ASYNC_LDS 1
#else
#define HAVE_ASYNC_LDS 0
#endif

// --- WMMA operand loaders (layouts per CDNA5 ISA 7.12.2, wave32) -----------
// A (16x32 bf16, row-major MxK source, leading dim ld):
//   lanes 0-15 : M=lane,    K chunks [k..k+7] and [k+16..k+23]
//   lanes16-31 : M=lane-16, K chunks [k+8..k+15] and [k+24..k+31]
__device__ __forceinline__ v16bf load_a_bf16(const __bf16* base, int ld) {
  const int lane = threadIdx.x & 31;
  const __bf16* p = base + (lane & 15) * ld + ((lane >> 4) << 3);
  union { v16bf v; f32x4 q[2]; } u;
  u.q[0] = *(const f32x4*)(p);
  u.q[1] = *(const f32x4*)(p + 16);
  return u.v;
}

// Same A layout but from an f32 source, converting to bf16 on the fly.
__device__ __forceinline__ v16bf load_a_f32(const float* base, int ld) {
  const int lane = threadIdx.x & 31;
  const float* p = base + (lane & 15) * ld + ((lane >> 4) << 3);
  v16bf r;
#pragma unroll
  for (int i = 0; i < 8; ++i) r[i] = (__bf16)p[i];
#pragma unroll
  for (int i = 0; i < 8; ++i) r[8 + i] = (__bf16)p[16 + i];
  return r;
}

// B (32x16 bf16). Source stored N-major: src[n*ld + k] == B[k][n].
//   lanes 0-15 : N=lane,    K chunks [0..7] and [16..23]
//   lanes16-31 : N=lane-16, K chunks [8..15] and [24..31]
__device__ __forceinline__ v16bf load_b_bf16(const __bf16* base, int ld) {
  const int lane = threadIdx.x & 31;
  const __bf16* p = base + (lane & 15) * ld + ((lane >> 4) << 3);
  union { v16bf v; f32x4 q[2]; } u;
  u.q[0] = *(const f32x4*)(p);
  u.q[1] = *(const f32x4*)(p + 16);
  return u.v;
}

// B operand from an LDS tile with padded leading dim LDP (ds_load_b128 x2,
// conflict-free: row stride = 68 dwords -> lane n covers banks 4n..4n+3).
__device__ __forceinline__ v16bf load_b_lds(const __bf16* base) {
  const int lane = threadIdx.x & 31;
  const __bf16* p = base + (lane & 15) * LDP + ((lane >> 4) << 3);
  union { v16bf v; f32x4 q[2]; } u;
  u.q[0] = *(const f32x4*)(p);
  u.q[1] = *(const f32x4*)(p + 16);
  return u.v;
}

__device__ __forceinline__ v8f wmma_bf16(v16bf a, v16bf b, v8f c) {
  // (neg_a, A, neg_b, B, c_mod, C, reuse_a, reuse_b)
  return __builtin_amdgcn_wmma_f32_16x16x32_bf16(false, a, false, b,
                                                 (short)0, c, false, false);
}

// Stage a 128x128 bf16 tile (row-major, ld=128) into LDS with ld=LDP.
// Uses the CDNA5 async Global->LDS DMA path when available (ASYNCcnt),
// else a plain b128 load + ds_store copy. Ends with a full block barrier.
__device__ __forceinline__ void stage128(const __bf16* __restrict__ g,
                                         __bf16* l) {
#if HAVE_ASYNC_LDS
  for (int c = threadIdx.x; c < 2048; c += blockDim.x) {
    int row = c >> 4, cc = (c & 15) << 3;        // 16 b128 chunks per row
    __builtin_amdgcn_global_load_async_to_lds_b128(
        (AS1 i32x4*)(g + row * 128 + cc),
        (AS3 i32x4*)(l + row * LDP + cc), 0, 0);
  }
  __builtin_amdgcn_s_wait_asynccnt(0);
#else
  for (int c = threadIdx.x; c < 2048; c += blockDim.x) {
    int row = c >> 4, cc = (c & 15) << 3;
    *(f32x4*)(l + row * LDP + cc) = *(const f32x4*)(g + row * 128 + cc);
  }
#endif
  __syncthreads();
}

// ---------------------------------------------------------------------------
// Kernel 0: transpose + convert the four 128x128 weights to bf16, N-major
// (WT[n*128 + k] = W[k*128 + n]) so they feed the WMMA B operand directly.
// ---------------------------------------------------------------------------
__global__ void k_prep_weights(const float* __restrict__ Wq,
                               const float* __restrict__ Wk,
                               const float* __restrict__ Wv,
                               const float* __restrict__ Wo,
                               __bf16* __restrict__ wT) {
  const float* W = (blockIdx.x == 0) ? Wq
                 : (blockIdx.x == 1) ? Wk
                 : (blockIdx.x == 2) ? Wv : Wo;
  __bf16* dst = wT + blockIdx.x * DD * DD;
  for (int o = threadIdx.x; o < DD * DD; o += blockDim.x) {
    int n = o >> 7, k = o & 127;
    dst[o] = (__bf16)W[k * DD + n];
  }
}

// ---------------------------------------------------------------------------
// Kernel 1: projection Y = X*W + b with head-split scatter.
// mode 0: out[((b*H+h)*Sp + sp)*128 + d]      (Q, K  : row-major per head)
// mode 1: out[((b*H+h)*128 + d)*Sp + sp]      (V     : transposed per head)
// Block = 256 threads = 8 waves; block tiles 128 rows, wave tiles 16x128.
// The 32 KB weight tile is staged once in LDS and shared by all 8 waves.
// ---------------------------------------------------------------------------
__global__ void k_proj(const float* __restrict__ X,
                       const __bf16* __restrict__ WT,
                       const float* __restrict__ bias,
                       __bf16* __restrict__ out, int mode) {
  __shared__ __bf16 wtile[128 * LDP];
  const int lane = threadIdx.x & 31;
  const int wave = threadIdx.x >> 5;
  const int r0   = blockIdx.x * 128 + wave * 16;

  stage128(WT, wtile);

  v8f acc[8];
#pragma unroll
  for (int t = 0; t < 8; ++t) acc[t] = (v8f)0.0f;

#pragma unroll
  for (int kk = 0; kk < 4; ++kk) {               // K = 128, 4 x 32
    v16bf a = load_a_f32(X + r0 * DD + kk * 32, DD);
#pragma unroll
    for (int nt = 0; nt < 8; ++nt) {
      v16bf b = load_b_lds(wtile + nt * 16 * LDP + kk * 32);
      acc[nt] = wmma_bf16(a, b, acc[nt]);
    }
  }

  const int mrow = (lane >> 4) << 3;             // 0 or 8
  int rb[8];
#pragma unroll
  for (int i = 0; i < 8; ++i) {
    int r  = r0 + mrow + i;
    int b_ = r >> 13;                            // / S
    int s  = r & (SS - 1);
    int sp = s >> 3, h = s & 7;                  // s = sp*H + h
    rb[i] = (mode == 0) ? (((b_ * HH + h) * SP + sp) * DD)
                        : (((b_ * HH + h) * DD) * SP + sp);
  }

#pragma unroll
  for (int nt = 0; nt < 8; ++nt) {
    int col = nt * 16 + (lane & 15);
    float bv = bias[col];
#pragma unroll
    for (int i = 0; i < 8; ++i) {
      float val = acc[nt][i] + bv;
      if (mode == 0) out[rb[i] + col]      = (__bf16)val;
      else           out[rb[i] + col * SP] = (__bf16)val;
    }
  }
}

// ---------------------------------------------------------------------------
// Kernel 2: scores = (Q Kᵀ)/sqrt(D) + mask*(-1e9) -> attention buffer (f32).
// Grid: bh(32) x qtile(8) x ktile(8) = 2048 blocks; wave tiles 16q x 128k.
// The 128x128 K-tile (shared by all 8 waves) is staged in LDS.
// ---------------------------------------------------------------------------
__global__ void k_scores(const __bf16* __restrict__ qp,
                         const __bf16* __restrict__ kp,
                         const float* __restrict__ mask,
                         float* __restrict__ attn) {
  __shared__ __bf16 ktile[128 * LDP];
  const int bid = blockIdx.x;
  const int kt  = bid & 7;
  const int qt  = (bid >> 3) & 7;
  const int bh  = bid >> 6;
  const int lane = threadIdx.x & 31;
  const int wave = threadIdx.x >> 5;

  const __bf16* qbase = qp + (bh * SP + qt * 128 + wave * 16) * DD;
  const __bf16* kbase = kp + (bh * SP + kt * 128) * DD;

  stage128(kbase, ktile);

  v8f acc[8];
#pragma unroll
  for (int t = 0; t < 8; ++t) acc[t] = (v8f)0.0f;

#pragma unroll
  for (int kk = 0; kk < 4; ++kk) {               // over D = 128
    v16bf a = load_a_bf16(qbase + kk * 32, DD);
#pragma unroll
    for (int nt = 0; nt < 8; ++nt) {
      v16bf b = load_b_lds(ktile + nt * 16 * LDP + kk * 32);
      acc[nt] = wmma_bf16(a, b, acc[nt]);
    }
  }

  const float scale = 0.08838834764831845f;      // 1/sqrt(128)
  const int mrow = (lane >> 4) << 3;
  const int rqb  = qt * 128 + wave * 16 + mrow;
  float* arow = attn + (long)bh * SP * SP;

#pragma unroll
  for (int nt = 0; nt < 8; ++nt) {
    int ck = kt * 128 + nt * 16 + (lane & 15);
#pragma unroll
    for (int i = 0; i < 8; ++i) {
      int rq = rqb + i;
      float v = acc[nt][i] * scale + mask[rq * SP + ck] * (-1e9f);
      arow[rq * SP + ck] = v;
    }
  }
}

// ---------------------------------------------------------------------------
// Kernel 3: row softmax in place over the 1024-wide attention rows.
// One 256-thread block per row; 4 f32 per thread; LDS tree reductions.
// ---------------------------------------------------------------------------
__global__ void k_softmax(float* __restrict__ attn) {
  __shared__ float red[256];
  float* p = attn + (long)blockIdx.x * SP;
  const int t = threadIdx.x;

  f32x4 v = *(const f32x4*)(p + t * 4);
  float m = fmaxf(fmaxf(v[0], v[1]), fmaxf(v[2], v[3]));
  red[t] = m; __syncthreads();
#pragma unroll
  for (int s = 128; s > 0; s >>= 1) {
    if (t < s) red[t] = fmaxf(red[t], red[t + s]);
    __syncthreads();
  }
  float rowmax = red[0]; __syncthreads();

  f32x4 e;
  e[0] = __expf(v[0] - rowmax); e[1] = __expf(v[1] - rowmax);
  e[2] = __expf(v[2] - rowmax); e[3] = __expf(v[3] - rowmax);
  red[t] = e[0] + e[1] + e[2] + e[3]; __syncthreads();
#pragma unroll
  for (int s = 128; s > 0; s >>= 1) {
    if (t < s) red[t] = red[t] + red[t + s];
    __syncthreads();
  }
  float inv = 1.0f / red[0];
  e[0] *= inv; e[1] *= inv; e[2] *= inv; e[3] *= inv;
  *(f32x4*)(p + t * 4) = e;
}

// ---------------------------------------------------------------------------
// Kernel 4: scaled = attn * V  (attn f32 -> bf16 on the fly, Vᵀ is bf16).
// Grid: bh(32) x qtile(8) = 256 blocks; K loop over 1024 keys (32 x 32).
// B tile changes every K step (8 KB, WGP$-resident) -> direct global loads.
// Writes directly into concat layout: concat[b, sp*H + h, d] (bf16).
// ---------------------------------------------------------------------------
__global__ void k_av(const float* __restrict__ attn,
                     const __bf16* __restrict__ vpT,
                     __bf16* __restrict__ concat) {
  const int bid = blockIdx.x;
  const int qt  = bid & 7;
  const int bh  = bid >> 3;
  const int b_  = bh >> 3, h = bh & 7;
  const int lane = threadIdx.x & 31;
  const int wave = threadIdx.x >> 5;

  const float*  abase = attn + ((long)bh * SP + qt * 128 + wave * 16) * SP;
  const __bf16* vbase = vpT + bh * DD * SP;

  v8f acc[8];
#pragma unroll
  for (int t = 0; t < 8; ++t) acc[t] = (v8f)0.0f;

  for (int kk = 0; kk < 32; ++kk) {              // over keys = 1024
    v16bf a = load_a_f32(abase + kk * 32, SP);
#pragma unroll
    for (int nt = 0; nt < 8; ++nt) {
      v16bf b = load_b_bf16(vbase + nt * 16 * SP + kk * 32, SP);
      acc[nt] = wmma_bf16(a, b, acc[nt]);
    }
  }

  const int mrow = (lane >> 4) << 3;
#pragma unroll
  for (int nt = 0; nt < 8; ++nt) {
    int d = nt * 16 + (lane & 15);
#pragma unroll
    for (int i = 0; i < 8; ++i) {
      int sp = qt * 128 + wave * 16 + mrow + i;
      int s  = sp * HH + h;
      concat[(b_ * SS + s) * DD + d] = (__bf16)acc[nt][i];
    }
  }
}

// ---------------------------------------------------------------------------
// Kernel 5: output = concat * Wo + bo  (f32 out, rows already in (b,s) order).
// Weight tile staged in LDS once per block.
// ---------------------------------------------------------------------------
__global__ void k_oproj(const __bf16* __restrict__ concat,
                        const __bf16* __restrict__ WoT,
                        const float* __restrict__ bo,
                        float* __restrict__ out) {
  __shared__ __bf16 wtile[128 * LDP];
  const int lane = threadIdx.x & 31;
  const int wave = threadIdx.x >> 5;
  const int r0   = blockIdx.x * 128 + wave * 16;

  stage128(WoT, wtile);

  v8f acc[8];
#pragma unroll
  for (int t = 0; t < 8; ++t) acc[t] = (v8f)0.0f;

#pragma unroll
  for (int kk = 0; kk < 4; ++kk) {
    v16bf a = load_a_bf16(concat + r0 * DD + kk * 32, DD);
#pragma unroll
    for (int nt = 0; nt < 8; ++nt) {
      v16bf b = load_b_lds(wtile + nt * 16 * LDP + kk * 32);
      acc[nt] = wmma_bf16(a, b, acc[nt]);
    }
  }

  const int mrow = (lane >> 4) << 3;
#pragma unroll
  for (int nt = 0; nt < 8; ++nt) {
    int col = nt * 16 + (lane & 15);
    float bv = bo[col];
#pragma unroll
    for (int i = 0; i < 8; ++i) {
      int r = r0 + mrow + i;
      out[r * DD + col] = acc[nt][i] + bv;
    }
  }
}

// ---------------------------------------------------------------------------
// Host launcher. Workspace layout (bytes):
//   [0,131072)          WqT/WkT/WvT/WoT bf16 (4 x 32 KB)
//   [131072, +8MB)      qp  bf16 (B,H,Sp,D)
//   [+8MB,  +8MB)       kp  bf16 (B,H,Sp,D)
//   [+8MB,  +8MB)       vpT bf16 (B,H,D,Sp)
//   [+8MB,  +8MB)       concat bf16 (B,S,D)
// Total ~33.7 MB.
// ---------------------------------------------------------------------------
extern "C" void kernel_launch(void* const* d_in, const int* in_sizes, int n_in,
                              void* d_out, int out_size, void* d_ws, size_t ws_size,
                              hipStream_t stream) {
  const float* v    = (const float*)d_in[0];
  const float* k    = (const float*)d_in[1];
  const float* q    = (const float*)d_in[2];
  const float* mask = (const float*)d_in[3];
  const float* Wq   = (const float*)d_in[4];
  const float* bq   = (const float*)d_in[5];
  const float* Wk   = (const float*)d_in[6];
  const float* bk   = (const float*)d_in[7];
  const float* Wv   = (const float*)d_in[8];
  const float* bv   = (const float*)d_in[9];
  const float* Wo   = (const float*)d_in[10];
  const float* bo   = (const float*)d_in[11];

  char* ws = (char*)d_ws;
  __bf16* wT     = (__bf16*)(ws);
  __bf16* qp     = (__bf16*)(ws + 131072);
  __bf16* kp     = (__bf16*)(ws + 131072 + 8388608);
  __bf16* vpT    = (__bf16*)(ws + 131072 + 2 * 8388608);
  __bf16* concat = (__bf16*)(ws + 131072 + 3 * 8388608);

  float* out  = (float*)d_out;
  float* attn = out + (size_t)BB * SS * DD;      // attention region of d_out

  k_prep_weights<<<4, 256, 0, stream>>>(Wq, Wk, Wv, Wo, wT);
  k_proj<<<256, 256, 0, stream>>>(q, wT + 0 * DD * DD, bq, qp, 0);
  k_proj<<<256, 256, 0, stream>>>(k, wT + 1 * DD * DD, bk, kp, 0);
  k_proj<<<256, 256, 0, stream>>>(v, wT + 2 * DD * DD, bv, vpT, 1);
  k_scores<<<2048, 256, 0, stream>>>(qp, kp, mask, attn);
  k_softmax<<<BB * HH * SP, 256, 0, stream>>>(attn);
  k_av<<<256, 256, 0, stream>>>(attn, vpT, concat);
  k_oproj<<<256, 256, 0, stream>>>(concat, wT + 3 * DD * DD, bo, out);
}
